// GatedLinearAttention_3582002724931
// MI455X (gfx1250) — matrix operand order repaired and checked by reading, not compiled, not run on hardware
//
#include <hip/hip_runtime.h>
#include <hip/hip_bf16.h>

// GLA forward for MI455X (gfx1250, wave32, WMMA).
// All matmuls on v_wmma_f32_16x16x32_bf16; operands staged in LDS as bf16 in
// fragment-contiguous layouts (fragment builds = ds_load_b128 pairs); all
// staging is vectorized (b128/b64 loads+stores, register-blocked transposes).

typedef __attribute__((ext_vector_type(16))) __bf16 v16bf;
typedef __attribute__((ext_vector_type(8)))  __bf16 v8bf;
typedef __attribute__((ext_vector_type(4)))  __bf16 v4bf;
typedef __attribute__((ext_vector_type(8)))  float  v8f;
typedef __attribute__((ext_vector_type(4)))  float  v4f;
typedef __attribute__((ext_vector_type(2)))  float  v2f;
typedef __attribute__((ext_vector_type(4)))  int    v4i;

#ifndef __has_builtin
#define __has_builtin(x) 0
#endif
#if __has_builtin(__builtin_amdgcn_global_load_async_to_lds_b128) && \
    __has_builtin(__builtin_amdgcn_s_wait_asynccnt)
#define HAVE_ASYNC_LDS 1
#else
#define HAVE_ASYNC_LDS 0
#endif

#define GLA_B 4
#define GLA_T 2048
#define GLA_D 1024
#define GLA_KD 1024
#define GLA_VD 2048
#define GLA_H 4
#define GLA_DK 256
#define GLA_DV 512
#define GLA_CHUNK 64
#define GLA_NCHUNK 32
#define GLA_ROWS (GLA_B * GLA_T) // 8192

// ---------------- WMMA fragment helpers (ISA 7.12.2 layouts) ----------------
// A fragment 16x32 bf16 from row-major [m][k] bf16 tile (ld elements).
// lane L: M=L&15, half=L>>4; element e -> K = (e<8 ? e : e+8) + 8*half.
__device__ __forceinline__ v16bf frag_a_bf(const __bf16* base, int ld, int lane) {
  int m = lane & 15, half = lane >> 4;
  const __bf16* p = base + m * ld + 8 * half;
  v8bf lo = *(const v8bf*)(p);
  v8bf hi = *(const v8bf*)(p + 16);
  return __builtin_shufflevector(lo, hi, 0, 1, 2, 3, 4, 5, 6, 7,
                                 8, 9, 10, 11, 12, 13, 14, 15);
}

// B fragment 32x16 bf16 from N-MAJOR [n][k] bf16 tile (ld elements).
// lane L: N=L&15, half=L>>4; element e -> K = (e<8 ? e : e+8) + 8*half.
__device__ __forceinline__ v16bf frag_b_bf(const __bf16* base, int ld, int lane) {
  int n = lane & 15, half = lane >> 4;
  const __bf16* p = base + n * ld + 8 * half;
  v8bf lo = *(const v8bf*)(p);
  v8bf hi = *(const v8bf*)(p + 16);
  return __builtin_shufflevector(lo, hi, 0, 1, 2, 3, 4, 5, 6, 7,
                                 8, 9, 10, 11, 12, 13, 14, 15);
}

// C/D f32 16x16: lane L (N=L&15); element i -> M = i + 8*(L>>4).
__device__ __forceinline__ void store_d_f32(float* base, int ms, int ns,
                                            int lane, v8f d) {
  int n = lane & 15, mb = 8 * (lane >> 4);
#pragma unroll
  for (int i = 0; i < 8; ++i) base[(size_t)(mb + i) * ms + n * ns] = d[i];
}

__device__ __forceinline__ void store_d_bf16(__bf16* base, int ld, int lane, v8f d) {
  int n = lane & 15, mb = 8 * (lane >> 4);
#pragma unroll
  for (int i = 0; i < 8; ++i) base[(mb + i) * ld + n] = (__bf16)d[i];
}

__device__ __forceinline__ v8f load_c_scaled(const float* base, int ms, int ns,
                                             int lane, const float* rowscale) {
  int n = lane & 15, mb = 8 * (lane >> 4);
  v8f c;
#pragma unroll
  for (int i = 0; i < 8; ++i)
    c[i] = base[(mb + i) * ms + n * ns] * rowscale[mb + i];
  return c;
}

#define WMMA_BF16(a, b, c) \
  __builtin_amdgcn_wmma_f32_16x16x32_bf16(false, (a), false, (b), (short)0, (c), false, false)

// ---------------- generic bf16-WMMA GEMM: C[MxN] = A[MxK] @ B[KxN] ----------
// block = 512 threads = 16 waves; block tile 128x64; K-step 64.
// Each wave owns one 16-row strip and two 16-col tiles (shared A fragment).
__global__ __launch_bounds__(512) void gemm_kernel(const float* __restrict__ A,
                                                   const float* __restrict__ B,
                                                   float* __restrict__ C,
                                                   int M, int N, int K) {
  __shared__ __bf16 sA[128 * 64]; // row-major [m][k]  16KB
  __shared__ __bf16 sBt[64 * 64]; // N-major   [n][k]   8KB
  int tid = threadIdx.x, lane = tid & 31, wave = tid >> 5;
  int bm = blockIdx.y * 128, bn = blockIdx.x * 64;
  int m0 = (wave & 7) * 16;
  int n0a = (wave >> 3) * 32, n0b = n0a + 16;
  v8f acc0 = {}, acc1 = {};
  // transpose block coords for sBt staging: 4 k-rows x 2 n-cols per thread
  int kb = (tid >> 5) * 4, nb = (tid & 31) * 2;
  for (int k0 = 0; k0 < K; k0 += 64) {
    // stage A: 2 runs of 8 contiguous elements per thread (b128 LDS stores)
#pragma unroll
    for (int it = 0; it < 2; ++it) {
      int i = tid * 8 + it * 512 * 8;
      int r = i >> 6, cc = i & 63;
      v4f a0 = *(const v4f*)&A[(size_t)(bm + r) * K + (k0 + cc)];
      v4f a1 = *(const v4f*)&A[(size_t)(bm + r) * K + (k0 + cc + 4)];
      v8bf pk;
#pragma unroll
      for (int j = 0; j < 4; ++j) { pk[j] = (__bf16)a0[j]; pk[4 + j] = (__bf16)a1[j]; }
      *(v8bf*)&sA[i] = pk;
    }
    // stage B transposed: 4x2 register block, b64 LDS stores along k
    {
      float c0[4], c1[4];
#pragma unroll
      for (int j = 0; j < 4; ++j) {
        v2f w = *(const v2f*)&B[(size_t)(k0 + kb + j) * N + (bn + nb)];
        c0[j] = w[0]; c1[j] = w[1];
      }
      v4bf p0, p1;
#pragma unroll
      for (int j = 0; j < 4; ++j) { p0[j] = (__bf16)c0[j]; p1[j] = (__bf16)c1[j]; }
      *(v4bf*)&sBt[nb * 64 + kb] = p0;
      *(v4bf*)&sBt[(nb + 1) * 64 + kb] = p1;
    }
    __syncthreads();
    if (k0 + 64 < K) { // gfx1250 global_prefetch_b8 of next K tiles
      __builtin_prefetch(&A[(size_t)(bm + (tid >> 2)) * K + (k0 + 64 + ((tid & 3) << 4))], 0, 1);
      __builtin_prefetch(&B[(size_t)(k0 + 64 + (tid >> 4)) * N + (bn + ((tid & 15) << 2))], 0, 1);
    }
#pragma unroll
    for (int kk = 0; kk < 64; kk += 32) {
      v16bf af = frag_a_bf(&sA[m0 * 64 + kk], 64, lane);
      acc0 = WMMA_BF16(af, frag_b_bf(&sBt[n0a * 64 + kk], 64, lane), acc0);
      acc1 = WMMA_BF16(af, frag_b_bf(&sBt[n0b * 64 + kk], 64, lane), acc1);
    }
    __syncthreads();
  }
  store_d_f32(&C[(size_t)(bm + m0) * N + (bn + n0a)], N, 1, lane, acc0);
  store_d_f32(&C[(size_t)(bm + m0) * N + (bn + n0b)], N, 1, lane, acc1);
}

// ---------------- low-rank gate: gk = log_sigmoid(hs@Wgk1@Wgk2+b)/16 --------
__global__ __launch_bounds__(256) void gk_kernel(const float* __restrict__ hs,
                                                 const float* __restrict__ Wgk1,
                                                 const float* __restrict__ Wgk2,
                                                 const float* __restrict__ bgk2,
                                                 float* __restrict__ gkout) {
  __shared__ float part[256];
  __shared__ float t1[16];
  int row = blockIdx.x;
  const float* x = hs + (size_t)row * GLA_D;
  int j = threadIdx.x & 15;
  int seg = threadIdx.x >> 4;
  float s = 0.f;
  for (int d = seg * 64; d < seg * 64 + 64; ++d) s += x[d] * Wgk1[d * 16 + j];
  part[threadIdx.x] = s;
  __syncthreads();
  if (threadIdx.x < 16) {
    float t = 0.f;
    for (int g = 0; g < 16; ++g) t += part[g * 16 + threadIdx.x];
    t1[threadIdx.x] = t;
  }
  __syncthreads();
  for (int c = threadIdx.x; c < GLA_KD; c += 256) {
    float acc = bgk2[c];
#pragma unroll
    for (int r = 0; r < 16; ++r) acc += t1[r] * Wgk2[r * GLA_KD + c];
    float ls = fminf(acc, 0.f) - log1pf(expf(-fabsf(acc))); // stable log-sigmoid
    gkout[(size_t)row * GLA_KD + c] = ls * (1.0f / 16.0f);
  }
}

// ---------------- chunked GLA recurrence ------------------------------------
// grid = B*H*(Dv/64) = 128 WGs; 512 threads (16 waves). Each WG owns a
// 256x64 f32 state slice (stored transposed S_t[dv][dk]) and scans 32 chunks.
__global__ __launch_bounds__(512) void gla_kernel(const float* __restrict__ qg_,
                                                  const float* __restrict__ kg_,
                                                  const float* __restrict__ gk_,
                                                  const float* __restrict__ vg_,
                                                  float* __restrict__ og_) {
  extern __shared__ char smem_raw[];
  float*  S_t    = (float*)smem_raw;                  // [dv=64][dk=256] 64KB
  float*  gbuf   = (float*)(smem_raw + 65536);        // [t=64][dk=256]  64KB
  float*  eglast = (float*)(smem_raw + 131072);       // [256]            1KB
  __bf16* qbuf   = (__bf16*)(smem_raw + 132096);      // [t][dk]         32KB
  __bf16* kbuf   = qbuf + 64 * 256;                   // [t][dk]         32KB
  __bf16* kbufT  = kbuf + 64 * 256;                   // [dk][t]*eglast  32KB
  __bf16* sbf    = kbufT + 256 * 64;                  // bf16(S_t)       32KB
  __bf16* vbufT  = sbf + 64 * 256;                    // [dv][t]          8KB
  __bf16* abuf   = vbufT + 64 * 64;                   // [tq][tk]         8KB

  int bid = blockIdx.x;
  int vs = bid & 7, h = (bid >> 3) & 3, b = bid >> 5;
  int tid = threadIdx.x, lane = tid & 31, wave = tid >> 5;
  int tr = wave & 3, tc = wave >> 2;
  int m0 = tr * 16, n0 = tc * 16;
  // transpose block coords (4 time-rows x 2 dv-cols per thread) for vbufT
  int tb = (tid >> 5) * 4, db = (tid & 31) * 2;

  for (int i = tid; i < 64 * 256; i += 512) S_t[i] = 0.f;
  __syncthreads();

  const int qkcol = h * GLA_DK;
  const int vcol  = h * GLA_DV + vs * 64;

  for (int c = 0; c < GLA_NCHUNK; ++c) {
    size_t row0 = (size_t)b * GLA_T + (size_t)c * GLA_CHUNK;

    // 1) stage gates (f32, async if available) + v slice transposed bf16
#if HAVE_ASYNC_LDS
#pragma unroll
    for (int i = tid * 4; i < 64 * 256; i += 512 * 4) {
      int r = i >> 8, d = i & 255;
      const float* src = &gk_[(row0 + r) * GLA_KD + qkcol + d];
      __builtin_amdgcn_global_load_async_to_lds_b128(
          (__attribute__((address_space(1))) v4i*)(v4i*)(void*)src,
          (__attribute__((address_space(3))) v4i*)(v4i*)(void*)(gbuf + i), 0, 0);
    }
#else
#pragma unroll
    for (int i = tid * 4; i < 64 * 256; i += 512 * 4) {
      int r = i >> 8, d = i & 255;
      *(v4f*)&gbuf[i] = *(const v4f*)&gk_[(row0 + r) * GLA_KD + qkcol + d];
    }
#endif
    { // v slice: 4x2 register-block transpose, b64 LDS stores along t
      float c0[4], c1[4];
#pragma unroll
      for (int j = 0; j < 4; ++j) {
        v2f w = *(const v2f*)&vg_[(row0 + tb + j) * GLA_VD + vcol + db];
        c0[j] = w[0]; c1[j] = w[1];
      }
      v4bf p0, p1;
#pragma unroll
      for (int j = 0; j < 4; ++j) { p0[j] = (__bf16)c0[j]; p1[j] = (__bf16)c1[j]; }
      *(v4bf*)&vbufT[db * 64 + tb] = p0;
      *(v4bf*)&vbufT[(db + 1) * 64 + tb] = p1;
    }
#if HAVE_ASYNC_LDS
    __builtin_amdgcn_s_wait_asynccnt(0);
#endif
    __syncthreads();

    // 2) in-chunk cumsum of gates along time, per dk column
    if (tid < 256) {
      float s = 0.f;
      for (int r = 0; r < 64; ++r) { s += gbuf[r * 256 + tid]; gbuf[r * 256 + tid] = s; }
      eglast[tid] = expf(s);
    }
    __syncthreads();

    // 3) build bf16 operands (vectorized): qg, kd, kg^T, bf16 state mirror
#pragma unroll
    for (int i = tid * 4; i < 64 * 256; i += 512 * 4) {
      int r = i >> 8, d = i & 255;
      size_t gi = (row0 + r) * GLA_KD + qkcol + d;
      v4f qv = *(const v4f*)&qg_[gi];
      v4f kv = *(const v4f*)&kg_[gi];
      v4f gv = *(const v4f*)&gbuf[i];
      v4f sv = *(const v4f*)&S_t[i];
      v4bf qp, kp, sp;
      float kdv[4];
#pragma unroll
      for (int j = 0; j < 4; ++j) {
        float e = expf(gv[j]);
        qp[j] = (__bf16)(qv[j] * e * 0.0625f); // scale = Dk^-0.5
        kdv[j] = kv[j] / e;                    // k * exp(-g)
        kp[j] = (__bf16)kdv[j];
        sp[j] = (__bf16)sv[j];
      }
      *(v4bf*)&qbuf[i] = qp;
      *(v4bf*)&kbuf[i] = kp;
      *(v4bf*)&sbf[i] = sp;
#pragma unroll
      for (int j = 0; j < 4; ++j)
        kbufT[(d + j) * 64 + r] = (__bf16)(kdv[j] * eglast[d + j]); // kg = kd*e^{g_last}
    }
    __syncthreads();

    // 4) A = qg @ kd^T (64x64, K=256), tril-masked; skip dead upper tiles
    {
      v8f acc = {};
      if (n0 <= m0 + 15) { // tile touches the lower triangle
#pragma unroll
        for (int k0 = 0; k0 < 256; k0 += 32) {
          v16bf af = frag_a_bf(&qbuf[m0 * 256 + k0], 256, lane);
          v16bf bf = frag_b_bf(&kbuf[n0 * 256 + k0], 256, lane); // lane n = key time
          acc = WMMA_BF16(af, bf, acc);
        }
        int n = n0 + (lane & 15), mb = m0 + 8 * (lane >> 4);
#pragma unroll
        for (int i2 = 0; i2 < 8; ++i2) if (n > mb + i2) acc[i2] = 0.f;
      }
      store_d_bf16(&abuf[m0 * 64 + n0], 64, lane, acc);
    }
    __syncthreads();

    // 5) o = qg @ S (K=256) + A @ v (K=64); stream tile to global
    {
      v8f acc = {};
#pragma unroll
      for (int k0 = 0; k0 < 256; k0 += 32) {
        v16bf af = frag_a_bf(&qbuf[m0 * 256 + k0], 256, lane);
        v16bf bf = frag_b_bf(&sbf[n0 * 256 + k0], 256, lane); // lane n = dv
        acc = WMMA_BF16(af, bf, acc);
      }
#pragma unroll
      for (int k0 = 0; k0 < 64; k0 += 32) {
        if (k0 <= m0 + 15) { // A block entirely masked beyond this
          v16bf af = frag_a_bf(&abuf[m0 * 64 + k0], 64, lane);
          v16bf bf = frag_b_bf(&vbufT[n0 * 64 + k0], 64, lane);
          acc = WMMA_BF16(af, bf, acc);
        }
      }
      store_d_f32(&og_[(row0 + m0) * GLA_VD + vcol + n0], GLA_VD, 1, lane, acc);
    }
    __syncthreads();

    // 6) S = S * e^{g_last}[dk] + kg^T @ v  (256x64 out, K=64; 4 tiles/wave)
#pragma unroll
    for (int t4 = 0; t4 < 4; ++t4) {
      int tile = wave * 4 + t4;
      int sm0 = (tile >> 2) * 16; // dk
      int sn0 = (tile & 3) * 16;  // dv
      // S_t is [dv][dk]: element (m=dk, n=dv) at S_t[n*256 + m]
      v8f acc = load_c_scaled(&S_t[sn0 * 256 + sm0], 1, 256, lane, &eglast[sm0]);
#pragma unroll
      for (int k0 = 0; k0 < 64; k0 += 32) {
        v16bf af = frag_a_bf(&kbufT[sm0 * 64 + k0], 64, lane); // pre-scaled kg^T
        v16bf bf = frag_b_bf(&vbufT[sn0 * 64 + k0], 64, lane);
        acc = WMMA_BF16(af, bf, acc);
      }
      store_d_f32(&S_t[sn0 * 256 + sm0], 1, 256, lane, acc);
    }
    __syncthreads();
  }
}

// ---------------- RMS-norm over Dv + SiLU gate (in place on o) --------------
__global__ __launch_bounds__(256) void gate_kernel(float* __restrict__ o,
                                                   const float* __restrict__ g,
                                                   const float* __restrict__ rms_w) {
  __shared__ float red[256];
  int h = blockIdx.x & 3;
  size_t row = blockIdx.x >> 2;
  float* orow = o + row * GLA_VD + (size_t)h * GLA_DV;
  const float* grow = g + row * GLA_VD + (size_t)h * GLA_DV;
  float x0 = orow[threadIdx.x], x1 = orow[threadIdx.x + 256];
  red[threadIdx.x] = x0 * x0 + x1 * x1;
  __syncthreads();
  for (int s = 128; s > 0; s >>= 1) {
    if (threadIdx.x < s) red[threadIdx.x] += red[threadIdx.x + s];
    __syncthreads();
  }
  float inv = rsqrtf(red[0] * (1.0f / GLA_DV) + 1e-5f);
#pragma unroll
  for (int e = 0; e < 2; ++e) {
    int dv = threadIdx.x + e * 256;
    float gt = grow[dv];
    float val = orow[dv] * inv * rms_w[dv];
    orow[dv] = val * (gt / (1.f + expf(-gt))); // gt * sigmoid(gt)
  }
}

// ---------------- launch ----------------------------------------------------
extern "C" void kernel_launch(void* const* d_in, const int* in_sizes, int n_in,
                              void* d_out, int out_size, void* d_ws, size_t ws_size,
                              hipStream_t stream) {
  const float* hs    = (const float*)d_in[0];
  const float* Wq    = (const float*)d_in[1];
  const float* Wk    = (const float*)d_in[2];
  const float* Wv    = (const float*)d_in[3];
  const float* Wg    = (const float*)d_in[4];
  const float* Wgk1  = (const float*)d_in[5];
  const float* Wgk2  = (const float*)d_in[6];
  const float* bgk2  = (const float*)d_in[7];
  const float* rms_w = (const float*)d_in[8];
  const float* Wo    = (const float*)d_in[9];
  float* out = (float*)d_out;

  // workspace (floats): q | k | v | g | gk | o  == 288MB total
  float* ws = (float*)d_ws;
  const size_t R = GLA_ROWS;
  float* q  = ws;
  float* k  = q  + R * GLA_KD;
  float* v  = k  + R * GLA_KD;
  float* g  = v  + R * GLA_VD;
  float* gk = g  + R * GLA_VD;
  float* o  = gk + R * GLA_KD;

  dim3 blk(512);
  gemm_kernel<<<dim3(GLA_KD / 64, R / 128), blk, 0, stream>>>(hs, Wq, q, R, GLA_KD, GLA_D);
  gemm_kernel<<<dim3(GLA_KD / 64, R / 128), blk, 0, stream>>>(hs, Wk, k, R, GLA_KD, GLA_D);
  gemm_kernel<<<dim3(GLA_VD / 64, R / 128), blk, 0, stream>>>(hs, Wv, v, R, GLA_VD, GLA_D);
  gemm_kernel<<<dim3(GLA_VD / 64, R / 128), blk, 0, stream>>>(hs, Wg, g, R, GLA_VD, GLA_D);
  gk_kernel<<<dim3(R), dim3(256), 0, stream>>>(hs, Wgk1, Wgk2, bgk2, gk);
  size_t smem = 132096 + (4 * 64 * 256 + 2 * 64 * 64) * sizeof(__bf16); // 273KB
  gla_kernel<<<dim3(GLA_B * GLA_H * (GLA_DV / 64)), blk, smem, stream>>>(q, k, gk, v, o);
  gate_kernel<<<dim3(R * GLA_H), dim3(256), 0, stream>>>(o, g, rms_w);
  gemm_kernel<<<dim3(GLA_D / 64, R / 128), blk, 0, stream>>>(o, Wo, out, R, GLA_D, GLA_VD);
}
